// NoHiT_46102178955524
// MI455X (gfx1250) — matrix operand order repaired and checked by reading, not compiled, not run on hardware
//
#include <hip/hip_runtime.h>

// ---------------------------------------------------------------------------
// NoHiT (tensor-ring product + 2x [3x3 conv + LeakyReLU]) for MI455X / gfx1250
//
// Stage 1: prep_A / prep_B : fold Z3*Z1 -> A[(i,k),(c,b)] bf16; Z2 -> Bt[j,(c,b)]
//          prep_W (x2)     : pre-swizzle conv weights into WMMA A-fragment
//                            order (bf16), one 32B vector load per fragment
// Stage 2: trp_gemm        : X = A x B via v_wmma_f32_16x16x32_bf16
//                            (M=32768, N=1024, K=256), stored bf16 HWC
//                            Xin[h=j][w=i][c=k3]  (pixel = 64 B contiguous)
// Stage 3: conv3x3_wmma<0> : conv1+bias+leaky -> bf16 HWC; input halo slab
//                            staged in LDS via global_load_async_to_lds_b128
// Stage 4: conv3x3_wmma<1> : conv2+bias+leaky -> f32 HWC (= reference
//                            (S2,S1,S3)), non-temporal stores (no reuse)
//
// Roofline: ~56 GFLOP vs ~320 MB of streams -> memory bound at 23.3 TB/s.
// bf16 activations + f32 WMMA accumulation; intermediates sized for L2.
// ---------------------------------------------------------------------------

typedef __bf16 bf16_t;
typedef __attribute__((ext_vector_type(16))) __bf16 v16bf;
typedef __attribute__((ext_vector_type(8)))  __bf16 v8bf;
typedef __attribute__((ext_vector_type(8)))  float  v8f;

#define CS1 1024
#define CS2 1024
#define CS3 32
#define CR  16

// ---------------------------------------------------------------------------
// A[(i*32+k3), (c*16+b)] = sum_a Z3[c,k3,a] * Z1[a,i,b]   (bf16 out)
// ---------------------------------------------------------------------------
__global__ void __launch_bounds__(256)
prep_A(const float* __restrict__ Z1, const float* __restrict__ Z3,
       bf16_t* __restrict__ Abig) {
  int idx = blockIdx.x * blockDim.x + threadIdx.x;   // 32768*256 threads
  int kk = idx & 255;            // c*16+b
  int m  = idx >> 8;             // i*32+k3
  int c = kk >> 4, b = kk & 15;
  int i = m >> 5,  k3 = m & 31;
  const float* z3 = Z3 + (c * CS3 + k3) * CR;        // Z3[c][k3][a], a stride 1
  const float* z1 = Z1 + i * CR + b;                 // Z1[a][i][b], a stride S1*R
  float s = 0.f;
#pragma unroll
  for (int a = 0; a < CR; ++a) s += z3[a] * z1[a * (CS1 * CR)];
  Abig[idx] = (bf16_t)s;
}

// ---------------------------------------------------------------------------
// Bt[j, (c*16+b)] = Z2[b, j, c]   (bf16, N-major: each lane's 16 K contiguous)
// ---------------------------------------------------------------------------
__global__ void __launch_bounds__(256)
prep_B(const float* __restrict__ Z2, bf16_t* __restrict__ Bt) {
  int idx = blockIdx.x * blockDim.x + threadIdx.x;   // 1024*256 threads
  int kk = idx & 255;
  int j  = idx >> 8;
  int c = kk >> 4, b = kk & 15;
  Bt[idx] = (bf16_t)Z2[(b * CS2 + j) * CR + c];
}

// ---------------------------------------------------------------------------
// Pre-swizzle OIHW f32 weights into per-lane WMMA A-fragments (bf16):
//   Wfrag[((t*2 + f)*32 + lane)*16 + e] = W[o=f*16+(lane&15)][c][ky][kx]
//   with c = ((lane>=16)?16:0) + e,  t = ky*3+kx.
// 9 taps * 2 frags * 32 lanes * 16 elems = 9216 bf16 (18 KB).
// ---------------------------------------------------------------------------
__global__ void __launch_bounds__(256)
prep_W(const float* __restrict__ W, bf16_t* __restrict__ Wfrag) {
  int idx = blockIdx.x * blockDim.x + threadIdx.x;   // 9216 threads
  if (idx >= 9 * 2 * 32 * 16) return;
  int e    = idx & 15;
  int lane = (idx >> 4) & 31;
  int f    = (idx >> 9) & 1;
  int t    = idx >> 10;
  int o  = f * 16 + (lane & 15);
  int kA = (lane >> 4) ? 16 : 0;
  int c  = kA + e;
  Wfrag[idx] = (bf16_t)W[((o * 32 + c) * 9) + t];
}

// ---------------------------------------------------------------------------
// WMMA GEMM, one wave per 16x16 tile, 8 K-steps of 32.
// D tile rows are consecutive k3 at fixed (i, j): pack 8 bf16 -> one 16B store
// into HWC image Xin[j][i][k3].
// ---------------------------------------------------------------------------
__global__ void __launch_bounds__(128)
trp_gemm(const bf16_t* __restrict__ Abig, const bf16_t* __restrict__ Bt,
         bf16_t* __restrict__ Xin) {
  const int wave = threadIdx.x >> 5;
  const int lane = threadIdx.x & 31;
  const int tile = blockIdx.x * 4 + wave;    // 2048 * 64 tiles
  const int tn = tile & 63;                  // N/16
  const int tm = tile >> 6;                  // M/16
  const int r  = lane & 15;
  const int hi = lane >> 4;                  // lane-group (K-half select)

  const bf16_t* arow = Abig + (tm * 16 + r) * 256 + (hi ? 16 : 0);
  const bf16_t* brow = Bt   + (tn * 16 + r) * 256 + (hi ? 16 : 0);

  v8f acc = {};
#pragma unroll
  for (int kc = 0; kc < 8; ++kc) {
    v8bf a_lo = *(const v8bf*)(arow + kc * 32);        // K = kc*32 + kb + 0..7
    v8bf a_hi = *(const v8bf*)(arow + kc * 32 + 8);    // K = kc*32 + kb + 8..15
    v16bf a = __builtin_shufflevector(a_lo, a_hi,
                0,1,2,3,4,5,6,7,8,9,10,11,12,13,14,15);
    v16bf bm = *(const v16bf*)(brow + kc * 32);        // 16 contiguous K
    acc = __builtin_amdgcn_wmma_f32_16x16x32_bf16(
              false, a, false, bm, (short)0, acc, false, false);
  }

  // m = tm*16 + v + 8*hi  ->  i = tm>>1 (const), k3 = (tm&1)*16 + hi*8 + v
  const int jcol = tn * 16 + r;
  const int i    = tm >> 1;
  const int kb   = ((tm & 1) << 4) + hi * 8;
  v8bf pack;
#pragma unroll
  for (int v = 0; v < 8; ++v) pack[v] = (bf16_t)acc[v];
  *(v8bf*)(Xin + (((size_t)jcol << 10) + i) * 32 + kb) = pack;
}

// ---------------------------------------------------------------------------
// 3x3 SAME conv, 32->32 ch, fused bias + LeakyReLU(0.01), HWC layouts.
// Block = 4 waves = 64 output px at fixed h. Input halo slab
// (3 rows x 66 px x 32 ch bf16, ~13 KB) staged in LDS via async-to-LDS DMA;
// B fragments from LDS; A fragments are single 32B loads from pre-swizzled
// weights. MODE 0: bf16 HWC out.  MODE 1: f32 HWC out, non-temporal.
// ---------------------------------------------------------------------------
template <int MODE>
__global__ void __launch_bounds__(128)
conv3x3_wmma(const bf16_t* __restrict__ xin, const bf16_t* __restrict__ Wfrag,
             const float* __restrict__ bias, void* __restrict__ outp) {
  __shared__ __align__(128) bf16_t lds[3 * 68 * 32];   // row stride 68 px

  const int tid  = threadIdx.x;
  const int wave = tid >> 5;
  const int lane = tid & 31;
  const int h  = blockIdx.y;
  const int w0 = blockIdx.x * 64;
  const int r  = lane & 15;
  const int hi = lane >> 4;

  // ---- stage halo slab: 3*66 px * 4 chunks of 16B; per-lane async DMA ----
  for (int q = tid; q < 3 * 66 * 4; q += 128) {
    const int c4  = q & 3;              // 16B chunk within pixel (8 bf16)
    const int pxi = (q >> 2) % 66;      // 0..65  (covers w0-1 .. w0+64)
    const int row = (q >> 2) / 66;      // 0..2   (covers h-1 .. h+1)
    const int hh = h + row - 1;
    const int ww = w0 + pxi - 1;
    const int le = (row * 68 + pxi) * 32 + c4 * 8;     // LDS element index
    if ((hh >= 0) & (hh < CS2) & (ww >= 0) & (ww < CS1)) {
      const bf16_t* g = xin + (((size_t)hh << 10) + ww) * 32 + c4 * 8;
      const unsigned loff = (unsigned)(unsigned long long)&lds[le];
      asm volatile("global_load_async_to_lds_b128 %0, %1, off"
                   :: "v"(loff), "v"(g) : "memory");
    } else {
      v8bf z = {};
      *(v8bf*)&lds[le] = z;             // zero-pad halo
    }
  }
  asm volatile("s_wait_asynccnt 0x0" ::: "memory");
  __syncthreads();

  v8f acc0 = {}, acc1 = {};
  const bf16_t* wl = Wfrag + lane * 16;          // this lane's fragment slot
#pragma unroll
  for (int t = 0; t < 9; ++t) {
    const int dy = t / 3, dx = t % 3;   // 0..2
    // B fragment: 16 contiguous channels at slab pixel (dy, wave*16+r+dx)
    const int px = wave * 16 + r + dx;
    const v16bf bm = *(const v16bf*)&lds[(dy * 68 + px) * 32 + hi * 16];

    // A fragments: one 32B vector load each (pre-swizzled bf16)
    const v16bf a0 = *(const v16bf*)(wl + (t * 2 + 0) * 512);
    const v16bf a1 = *(const v16bf*)(wl + (t * 2 + 1) * 512);

    acc0 = __builtin_amdgcn_wmma_f32_16x16x32_bf16(
               false, a0, false, bm, (short)0, acc0, false, false);
    acc1 = __builtin_amdgcn_wmma_f32_16x16x32_bf16(
               false, a1, false, bm, (short)0, acc1, false, false);
  }

  // ---- epilogue: bias + leaky, packed vector stores (HWC) ----
  const int w = w0 + wave * 16 + r;
  const v8f bv0 = *(const v8f*)(bias + hi * 8);        // channels hi*8..+7
  const v8f bv1 = *(const v8f*)(bias + hi * 8 + 16);   // channels 16+hi*8..+7
  if (MODE == 0) {
    bf16_t* o = (bf16_t*)outp + (((size_t)h << 10) + w) * 32;
    v8bf p0, p1;
#pragma unroll
    for (int v = 0; v < 8; ++v) {
      float a = acc0[v] + bv0[v];
      float b = acc1[v] + bv1[v];
      a = (a >= 0.f) ? a : 0.01f * a;
      b = (b >= 0.f) ? b : 0.01f * b;
      p0[v] = (bf16_t)a;
      p1[v] = (bf16_t)b;
    }
    *(v8bf*)(o + hi * 8)      = p0;
    *(v8bf*)(o + hi * 8 + 16) = p1;
  } else {
    float* o = (float*)outp + (((size_t)h << 10) + w) * 32;
    v8f q0, q1;
#pragma unroll
    for (int v = 0; v < 8; ++v) {
      float a = acc0[v] + bv0[v];
      float b = acc1[v] + bv1[v];
      q0[v] = (a >= 0.f) ? a : 0.01f * a;
      q1[v] = (b >= 0.f) ? b : 0.01f * b;
    }
    // final output: never re-read -> keep it out of L2
    __builtin_nontemporal_store(q0, (v8f*)(o + hi * 8));
    __builtin_nontemporal_store(q1, (v8f*)(o + hi * 8 + 16));
  }
}

// ---------------------------------------------------------------------------
extern "C" void kernel_launch(void* const* d_in, const int* in_sizes, int n_in,
                              void* d_out, int out_size, void* d_ws, size_t ws_size,
                              hipStream_t stream) {
  const float* Z1 = (const float*)d_in[0];
  const float* Z2 = (const float*)d_in[1];
  const float* Z3 = (const float*)d_in[2];
  const float* W1 = (const float*)d_in[3];
  const float* b1 = (const float*)d_in[4];
  const float* W2 = (const float*)d_in[5];
  const float* b2 = (const float*)d_in[6];

  char* ws = (char*)d_ws;
  bf16_t* Abig = (bf16_t*)(ws);                          // 16 MB
  bf16_t* Bt   = (bf16_t*)(ws + (16u  << 20));           // 0.5 MB
  bf16_t* Xin  = (bf16_t*)(ws + (17u  << 20));           // HWC bf16: 64 MB
  bf16_t* Y1   = (bf16_t*)(ws + (81u  << 20));           // HWC bf16: 64 MB
  bf16_t* Wf1  = (bf16_t*)(ws + (145u << 20));           // 18 KB
  bf16_t* Wf2  = (bf16_t*)(ws + (145u << 20) + 32768);   // 18 KB

  prep_A<<<32768, 256, 0, stream>>>(Z1, Z3, Abig);
  prep_B<<<1024,  256, 0, stream>>>(Z2, Bt);
  prep_W<<<36,    256, 0, stream>>>(W1, Wf1);
  prep_W<<<36,    256, 0, stream>>>(W2, Wf2);

  trp_gemm<<<32768, 128, 0, stream>>>(Abig, Bt, Xin);

  dim3 cgrid(16, 1024, 1);   // 16 blocks x 64 px cover each of 1024 rows
  conv3x3_wmma<0><<<cgrid, 128, 0, stream>>>(Xin, Wf1, b1, (void*)Y1);
  conv3x3_wmma<1><<<cgrid, 128, 0, stream>>>(Y1,  Wf2, b2, d_out);
}
